// GCNModel_4148938408550
// MI455X (gfx1250) — matrix operand order repaired and checked by reading, not compiled, not run on hardware
//
#include <hip/hip_runtime.h>
#include <hip/hip_bf16.h>

typedef __attribute__((ext_vector_type(16))) __bf16 v16bf;
typedef __attribute__((ext_vector_type(8)))  float  v8f;

#define F_IN  128
#define HID   256
#define NCLS  32

static __device__ __forceinline__ __bf16 f2bf(float f) { return (__bf16)f; }

static __device__ __forceinline__ void atomic_add_f32(float* p, float v) {
    __hip_atomic_fetch_add(p, v, __ATOMIC_RELAXED, __HIP_MEMORY_SCOPE_AGENT);
}

// ---------------------------------------------------------------- degree / norm
__global__ __launch_bounds__(256) void k_init_deg(float* deg, int n) {
    int i = blockIdx.x * blockDim.x + threadIdx.x;
    if (i < n) deg[i] = 1.0f;               // self-loop contributes 1 to every node
}

__global__ __launch_bounds__(256) void k_acc_deg(const long long* __restrict__ ei,
                                                 float* deg, int E) {
    int e = blockIdx.x * blockDim.x + threadIdx.x;
    if (e >= E) return;
    long long d = ei[(size_t)E + e];        // dst row of edge_index
    atomic_add_f32(&deg[d], 1.0f);
}

__global__ __launch_bounds__(256) void k_dinv(float* deg, int n) {
    int i = blockIdx.x * blockDim.x + threadIdx.x;
    if (i >= n) return;
    float dg = deg[i];
    deg[i] = dg > 0.0f ? rsqrtf(dg) : 0.0f; // in-place: deg -> dinv
}

// ---------------------------------------------------------------- weight prep (bf16, transposed [out][in])
__global__ __launch_bounds__(256) void k_cvt_w1(const float* __restrict__ W1, __bf16* __restrict__ W1t) {
    int i = blockIdx.x * blockDim.x + threadIdx.x;   // i over HID*F_IN
    if (i >= HID * F_IN) return;
    int ncol = i / F_IN, k = i % F_IN;
    W1t[i] = f2bf(W1[(size_t)k * HID + ncol]);       // W1 is [F_IN][HID] row-major
}

__global__ __launch_bounds__(256) void k_cvt_w2(const float* __restrict__ W2, __bf16* __restrict__ W2t) {
    int i = blockIdx.x * blockDim.x + threadIdx.x;   // i over NCLS*HID
    if (i >= NCLS * HID) return;
    int ncol = i / HID, k = i % HID;
    W2t[i] = f2bf(W2[(size_t)k * NCLS + ncol]);      // W2 is [HID][NCLS] row-major
}

// ---------------------------------------------------------------- propagation layer 1 (128-wide)
__global__ __launch_bounds__(256) void k_prop1_init(const float* __restrict__ x,
                                                    const float* __restrict__ dinv,
                                                    float* __restrict__ p, int n) {
    int i = blockIdx.x * blockDim.x + threadIdx.x;   // over n*F_IN
    if (i >= n * F_IN) return;
    int node = i / F_IN;
    float dv = dinv[node];
    p[i] = dv * dv * x[i];                           // self-loop term
}

__global__ __launch_bounds__(256) void k_prop1_edges(const float* __restrict__ x,
                                                     const long long* __restrict__ ei,
                                                     const float* __restrict__ dinv,
                                                     float* __restrict__ p, int E) {
    int wave = blockIdx.x * (blockDim.x >> 5) + (threadIdx.x >> 5);
    int lane = threadIdx.x & 31;
    if (wave >= E) return;
    long long s = ei[wave];
    long long d = ei[(size_t)E + wave];
    float ns = dinv[s] * dinv[d];
    const float4 xv = *(const float4*)(x + (size_t)s * F_IN + lane * 4);
    float* pd = p + (size_t)d * F_IN + lane * 4;
    atomic_add_f32(pd + 0, ns * xv.x);
    atomic_add_f32(pd + 1, ns * xv.y);
    atomic_add_f32(pd + 2, ns * xv.z);
    atomic_add_f32(pd + 3, ns * xv.w);
}

// ---------------------------------------------------------------- GEMM1: h1 = relu(p @ W1 + b1), bf16 out
// one wave per 16-row M-tile; loops all 16 N-tiles reusing the A fragments
__global__ __launch_bounds__(256) void k_gemm1_wmma(const float* __restrict__ p,
                                                    const __bf16* __restrict__ W1t,
                                                    const float* __restrict__ b1,
                                                    __bf16* __restrict__ h1, int M) {
    int mtile = blockIdx.x * (blockDim.x >> 5) + (threadIdx.x >> 5);
    if (mtile * 16 >= M) return;
    int lane = threadIdx.x & 31;
    int hi   = lane >> 4;         // lane group (0: lanes 0-15, 1: lanes 16-31)
    int mrow = mtile * 16 + (lane & 15);
    int colL = lane & 15;

    // A fragments: 16x32 bf16 per K-step, K=128 -> 4 fragments (contiguous K order: element j holds k = hi*16 + j)
    v16bf afrag[4];
    const float* prow = p + (size_t)mrow * F_IN;
#pragma unroll
    for (int ks = 0; ks < 4; ++ks) {
        const float* ap = prow + ks * 32 + hi * 16;
        v16bf a;
#pragma unroll
        for (int j = 0; j < 8; ++j) {
            a[j]     = f2bf(ap[j]);          // K = ks*32 + hi*16 + j
            a[j + 8] = f2bf(ap[8 + j]);      // K = ks*32 + hi*16 + 8 + j
        }
        afrag[ks] = a;
    }

    for (int nt = 0; nt < HID / 16; ++nt) {
        int ncol = nt * 16 + colL;
        const __bf16* wb = W1t + (size_t)ncol * F_IN + hi * 16;  // B: lane holds 16 contiguous K
        v8f c = {};
#pragma unroll
        for (int ks = 0; ks < 4; ++ks) {
            v16bf b;
#pragma unroll
            for (int j = 0; j < 16; ++j) b[j] = wb[ks * 32 + j]; // K = ks*32 + hi*16 + j
            c = __builtin_amdgcn_wmma_f32_16x16x32_bf16(false, afrag[ks], false, b,
                                                        (short)0, c, false, false);
        }
        float bias = b1[ncol];
#pragma unroll
        for (int r = 0; r < 8; ++r) {                 // D: M = hi*8 + r, N = colL
            int orow = mtile * 16 + hi * 8 + r;
            float v = c[r] + bias;
            v = v > 0.0f ? v : 0.0f;                  // ReLU fused
            h1[(size_t)orow * HID + ncol] = f2bf(v);
        }
    }
}

// ---------------------------------------------------------------- GEMM2: g = h1 @ W2, f32 out
__global__ __launch_bounds__(256) void k_gemm2_wmma(const __bf16* __restrict__ h1,
                                                    const __bf16* __restrict__ W2t,
                                                    float* __restrict__ g, int M) {
    int mtile = blockIdx.x * (blockDim.x >> 5) + (threadIdx.x >> 5);
    if (mtile * 16 >= M) return;
    int lane = threadIdx.x & 31;
    int hi   = lane >> 4;
    int mrow = mtile * 16 + (lane & 15);
    int colL = lane & 15;
    const __bf16* hrow = h1 + (size_t)mrow * HID;

    for (int nt = 0; nt < NCLS / 16; ++nt) {
        int ncol = nt * 16 + colL;
        const __bf16* wb = W2t + (size_t)ncol * HID + hi * 16;
        v8f c = {};
#pragma unroll
        for (int ks = 0; ks < HID / 32; ++ks) {
            const __bf16* ap = hrow + ks * 32 + hi * 16;
            v16bf a, b;
#pragma unroll
            for (int j = 0; j < 8; ++j) { a[j] = ap[j]; a[j + 8] = ap[8 + j]; }
#pragma unroll
            for (int j = 0; j < 16; ++j) b[j] = wb[ks * 32 + j];
            c = __builtin_amdgcn_wmma_f32_16x16x32_bf16(false, a, false, b,
                                                        (short)0, c, false, false);
        }
#pragma unroll
        for (int r = 0; r < 8; ++r) {
            int orow = mtile * 16 + hi * 8 + r;
            g[(size_t)orow * NCLS + ncol] = c[r];
        }
    }
}

// ---------------------------------------------------------------- propagation layer 2 (32-wide)
__global__ __launch_bounds__(256) void k_prop2_init(const float* __restrict__ g,
                                                    const float* __restrict__ dinv,
                                                    float* __restrict__ out, int n) {
    int i = blockIdx.x * blockDim.x + threadIdx.x;   // over n*NCLS
    if (i >= n * NCLS) return;
    int node = i / NCLS;
    float dv = dinv[node];
    out[i] = dv * dv * g[i];
}

__global__ __launch_bounds__(256) void k_prop2_edges(const float* __restrict__ g,
                                                     const long long* __restrict__ ei,
                                                     const float* __restrict__ dinv,
                                                     float* __restrict__ out, int E) {
    int wave = blockIdx.x * (blockDim.x >> 5) + (threadIdx.x >> 5);
    int lane = threadIdx.x & 31;                     // lane == class index (NCLS==32)
    if (wave >= E) return;
    long long s = ei[wave];
    long long d = ei[(size_t)E + wave];
    float ns = dinv[s] * dinv[d];
    float v = g[(size_t)s * NCLS + lane];
    atomic_add_f32(out + (size_t)d * NCLS + lane, ns * v);
}

// ---------------------------------------------------------------- bias + log_softmax (wave per row)
__global__ __launch_bounds__(256) void k_logsoftmax(float* __restrict__ out,
                                                    const float* __restrict__ b2, int n) {
    int row  = blockIdx.x * (blockDim.x >> 5) + (threadIdx.x >> 5);
    int lane = threadIdx.x & 31;
    if (row >= n) return;
    float v = out[(size_t)row * NCLS + lane] + b2[lane];
    float m = v;
#pragma unroll
    for (int off = 16; off > 0; off >>= 1) m = fmaxf(m, __shfl_xor(m, off, 32));
    float e = __expf(v - m);
    float s = e;
#pragma unroll
    for (int off = 16; off > 0; off >>= 1) s += __shfl_xor(s, off, 32);
    out[(size_t)row * NCLS + lane] = v - m - __logf(s);
}

// ================================================================ launch
extern "C" void kernel_launch(void* const* d_in, const int* in_sizes, int n_in,
                              void* d_out, int out_size, void* d_ws, size_t ws_size,
                              hipStream_t stream) {
    const float*     x  = (const float*)d_in[0];
    const long long* ei = (const long long*)d_in[1];   // int64 edge_index [2][E]
    const float*     W1 = (const float*)d_in[2];
    const float*     b1 = (const float*)d_in[3];
    const float*     W2 = (const float*)d_in[4];
    const float*     b2 = (const float*)d_in[5];
    float* out = (float*)d_out;

    const int N = in_sizes[0] / F_IN;                  // 100000
    const int E = in_sizes[1] / 2;                     // 1600000

    // workspace carve-up (256B aligned)
    char* ws = (char*)d_ws;
    size_t o = 0;
    auto al = [](size_t v) { return (v + 255) & ~(size_t)255; };
    float*  dinv = (float*)(ws + o);  o = al(o + (size_t)N * 4);
    float*  p    = (float*)(ws + o);  o = al(o + (size_t)N * F_IN * 4);
    __bf16* h1   = (__bf16*)(ws + o); o = al(o + (size_t)N * HID * 2);
    float*  g    = (float*)(ws + o);  o = al(o + (size_t)N * NCLS * 4);
    __bf16* W1t  = (__bf16*)(ws + o); o = al(o + (size_t)HID * F_IN * 2);
    __bf16* W2t  = (__bf16*)(ws + o); o = al(o + (size_t)NCLS * HID * 2);
    (void)ws_size; (void)n_in; (void)out_size;

    const int TB = 256;
    const int wavesPerBlk = TB / 32;

    // degrees -> dinv (in place)
    k_init_deg<<<(N + TB - 1) / TB, TB, 0, stream>>>(dinv, N);
    k_acc_deg<<<(E + TB - 1) / TB, TB, 0, stream>>>(ei, dinv, E);
    k_dinv<<<(N + TB - 1) / TB, TB, 0, stream>>>(dinv, N);

    // weights to bf16, transposed [out][in]
    k_cvt_w1<<<(HID * F_IN + TB - 1) / TB, TB, 0, stream>>>(W1, W1t);
    k_cvt_w2<<<(NCLS * HID + TB - 1) / TB, TB, 0, stream>>>(W2, W2t);

    // layer 1: p = A_hat @ x  (propagate 128-wide input first: halves scatter traffic)
    k_prop1_init<<<((size_t)N * F_IN + TB - 1) / TB, TB, 0, stream>>>(x, dinv, p, N);
    k_prop1_edges<<<(E + wavesPerBlk - 1) / wavesPerBlk, TB, 0, stream>>>(x, ei, dinv, p, E);

    // h1 = relu(p @ W1 + b1)  -- WMMA bf16
    int mtiles = N / 16;  // N divisible by 16
    k_gemm1_wmma<<<(mtiles + wavesPerBlk - 1) / wavesPerBlk, TB, 0, stream>>>(p, W1t, b1, h1, N);

    // g = h1 @ W2             -- WMMA bf16 (GEMM before propagation: 32-wide scatter)
    k_gemm2_wmma<<<(mtiles + wavesPerBlk - 1) / wavesPerBlk, TB, 0, stream>>>(h1, W2t, g, N);

    // layer 2 propagation into d_out, then bias + log_softmax in place
    k_prop2_init<<<((size_t)N * NCLS + TB - 1) / TB, TB, 0, stream>>>(g, dinv, out, N);
    k_prop2_edges<<<(E + wavesPerBlk - 1) / wavesPerBlk, TB, 0, stream>>>(g, ei, dinv, out, E);
    k_logsoftmax<<<(N + wavesPerBlk - 1) / wavesPerBlk, TB, 0, stream>>>(out, b2, N);
}
